// TGCNmodel_10591389352452
// MI455X (gfx1250) — compile-verified, matrix operand order checked by reading, not run on hardware
//
#include <hip/hip_runtime.h>
#include <hip/hip_bf16.h>
#include <math.h>

typedef _Float16 h16;
typedef __attribute__((ext_vector_type(16))) _Float16 v16h;
typedef __attribute__((ext_vector_type(8)))  _Float16 v8h;
typedef __attribute__((ext_vector_type(8)))  float    v8f;

#define NN    1000
#define EE    2000
#define BB    32
#define TT    64
#define HRR   64
#define HTT   256
#define INJND 753
#define BNN   (BB*NN)          // 32000
#define ETOT  (BB*EE + BNN)    // 96000
#define FLOODV 5.0f

// ---------------- device helpers ----------------

__device__ __forceinline__ float lrelu(float x, float s){ return x > 0.f ? x : s*x; }
__device__ __forceinline__ float sigf(float x){ return 1.f/(1.f+expf(-x)); }

// monotone float<->uint map so atomicMax(u32) == float max
__device__ __forceinline__ unsigned fflip(float f){
  unsigned u = __float_as_uint(f);
  return (u & 0x80000000u) ? ~u : (u | 0x80000000u);
}
__device__ __forceinline__ float funflip(unsigned u){
  u = (u & 0x80000000u) ? (u & 0x7FFFFFFFu) : ~u;
  return __uint_as_float(u);
}

// Load one 16-half WMMA fragment from a row-major [row][Kpad] f16 buffer.
// ISA 16-bit A layout: lane half=lane>>4; elems 0..7 at k=kb+half*8, elems 8..15 at k=kb+16+half*8.
__device__ __forceinline__ v16h load_frag(const h16* rowBase, int kb, int halfw){
  const v8h* p0 = (const v8h*)(rowBase + kb + halfw*8);
  const v8h* p1 = (const v8h*)(rowBase + kb + 16 + halfw*8);
  v8h lo = *p0, hi = *p1;
  v16h r;
#pragma unroll
  for (int i = 0; i < 8; ++i){ r[i] = lo[i]; r[i+8] = hi[i]; }
  return r;
}

// ---------------- one-time kernels ----------------

__global__ void k_build_edges(const int* ei, int* srcAll, int* dstAll){
  int e = blockIdx.x*blockDim.x + threadIdx.x;
  if (e >= ETOT) return;
  if (e < BB*EE){
    int b = e / EE, k = e % EE;
    srcAll[e] = ei[k]      + b*NN;
    dstAll[e] = ei[EE + k] + b*NN;
  } else {
    int n = e - BB*EE;
    srcAll[e] = n; dstAll[e] = n;   // self loops
  }
}

__global__ void k_f32_to_f16_pad(const float* src, h16* dst, int rows, int cols,
                                 int ld, int rowsPad){
  int i = blockIdx.x*blockDim.x + threadIdx.x;
  int tot = rowsPad * ld;
  if (i >= tot) return;
  int r = i / ld, c = i % ld;
  float v = (r < rows && c < cols) ? src[(size_t)r*cols + c] : 0.f;
  dst[i] = (h16)v;
}

__global__ void k_zero_f32(float* p, int n){
  int i = blockIdx.x*blockDim.x + threadIdx.x;
  if (i < n) p[i] = 0.f;
}

// LSTM: 1 block, 256 threads. h/c/gates live in LDS; hs written as f16 for the fc WMMA GEMM.
__global__ void k_lstm(const float* rain, const float* Wih, const float* Whh,
                       const float* bih, const float* bhh, h16* hs16){
  __shared__ float hS[BB][HRR];
  __shared__ float cS[BB][HRR];
  __shared__ float gS[BB][4*HRR];
  int tid = threadIdx.x;
  for (int i = tid; i < BB*HRR; i += 256){ hS[i/HRR][i%HRR] = 0.f; cS[i/HRR][i%HRR] = 0.f; }
  __syncthreads();
  float wih = Wih[tid];
  float bsum = bih[tid] + bhh[tid];
  const float* wr = Whh + tid*HRR;
  for (int t = 0; t < TT; ++t){
    // phase A: gate row `tid` for every batch
    for (int b = 0; b < BB; ++b){
      float acc = wih * rain[b*TT + t] + bsum;
#pragma unroll 8
      for (int k = 0; k < HRR; ++k) acc += wr[k] * hS[b][k];
      gS[b][tid] = acc;
    }
    __syncthreads();
    // phase B: cell/state update (2048 (b,u) pairs)
    for (int r = 0; r < 8; ++r){
      int idx = tid + 256*r;
      int b = idx / HRR, u = idx % HRR;
      float ig = gS[b][u], fg = gS[b][HRR+u], gg = gS[b][2*HRR+u], og = gS[b][3*HRR+u];
      float c = sigf(fg)*cS[b][u] + sigf(ig)*tanhf(gg);
      float h = sigf(og)*tanhf(c);
      cS[b][u] = c; hS[b][u] = h;
      hs16[(size_t)(t*BB + b)*HRR + u] = (h16)h;
    }
    __syncthreads();
  }
}

// ---------------- generic WMMA GEMM ----------------
// out[m, j] = act( sum_k A16[m][k] * B16[j][k] + bias[j] )
// A16: [Mtiles*16][lda] f16 row-major, K padded (mult of 32, zero-filled).
// B16: weight in native [out][in] layout, f16, K/N padded.
// act: 0 none, 1 leaky_relu(0.01), 2 softplus. One wave per 16x16 tile.
__global__ void k_wmma_gemm(const h16* __restrict__ A, int lda,
                            const h16* __restrict__ Bw, int ldb,
                            const float* __restrict__ bias,
                            int Mtiles, int Ntiles, int K, int actN, int act,
                            float* o1, int ldo1,
                            float* o2, int ldo2,
                            h16* oh, int ldoh){
  int wid  = threadIdx.x >> 5;
  int lane = threadIdx.x & 31;
  int tile = blockIdx.x * (blockDim.x >> 5) + wid;
  if (tile >= Mtiles*Ntiles) return;          // wave-uniform
  int tm = tile / Ntiles, tn = tile % Ntiles;
  int halfw = lane >> 4, lr = lane & 15;
  const h16* aRow = A  + (size_t)(tm*16 + lr)*lda;
  const h16* bRow = Bw + (size_t)(tn*16 + lr)*ldb;
  v8f acc = {};
  for (int kb = 0; kb < K; kb += 32){
    __builtin_prefetch(aRow + kb + 32, 0, 1);   // global_prefetch_b8
    __builtin_prefetch(bRow + kb + 32, 0, 1);
    v16h a = load_frag(aRow, kb, halfw);
    v16h b = load_frag(bRow, kb, halfw);
    acc = __builtin_amdgcn_wmma_f32_16x16x32_f16(
        /*neg_a=*/false, a, /*neg_b=*/false, b,
        /*c_mod=*/(short)0, acc, /*reuse_a=*/false, /*reuse_b=*/false);
  }
  int col = tn*16 + lr;
  if (col >= actN) return;                     // padded N columns: drop (uniform per lane-col)
  float bv = bias ? bias[col] : 0.f;
#pragma unroll
  for (int r = 0; r < 8; ++r){
    int m = tm*16 + r + halfw*8;               // D layout: VGPR r -> M=r (lanes 0-15), M=8+r (16-31)
    float v = acc[r] + bv;
    if (act == 1)      v = v > 0.f ? v : 0.01f*v;
    else if (act == 2) v = (v > 20.f) ? v : logf(1.f + expf(v));
    if (o1) o1[(size_t)m*ldo1 + col] = v;
    if (o2) o2[(size_t)m*ldo2 + col] = v;
    if (oh) oh[(size_t)m*ldoh + col] = (h16)v;
  }
}

// ---------------- per-step kernels ----------------

__global__ void k_prep1(const float* runoff, const float* inflow, const float* xn,
                        const float* g1w, const float* g1as, const float* g1ad,
                        int t, float* lateralOut, float* xbuf, unsigned char* status,
                        float* hg1, float* asrc1, float* adst1,
                        unsigned* m1, float* s1, float* out1){
  int n = blockIdx.x*blockDim.x + threadIdx.x;
  if (n >= BNN) return;
  int b = n / NN, i = n % NN;
  float inb = inflow[b*TT + t];
  float r = runoff[(size_t)(t*BB + b)*NN + i] + (i == INJND ? inb : 0.f);
  lateralOut[(size_t)b*(TT*NN) + (size_t)t*NN + i] = r;
  float x0 = xn[n*2+0], x1 = xn[n*2+1], x2 = r;
  xbuf[n*3+0]=x0; xbuf[n*3+1]=x1; xbuf[n*3+2]=x2;
  status[n] = (unsigned char)((x0 > FLOODV) || (inb > FLOODV));
  float hloc[24];
#pragma unroll
  for (int hc = 0; hc < 24; ++hc){
    float v = g1w[hc*3+0]*x0 + g1w[hc*3+1]*x1 + g1w[hc*3+2]*x2;
    hloc[hc] = v; hg1[(size_t)n*24 + hc] = v;
  }
#pragma unroll
  for (int h = 0; h < 8; ++h){
    float as = 0.f, ad = 0.f;
#pragma unroll
    for (int c = 0; c < 3; ++c){ as += hloc[h*3+c]*g1as[h*3+c]; ad += hloc[h*3+c]*g1ad[h*3+c]; }
    asrc1[n*8+h] = as; adst1[n*8+h] = ad;
    m1[n*8+h] = fflip(-3.0e38f);
    s1[n*8+h] = 0.f;
  }
#pragma unroll
  for (int hc = 0; hc < 24; ++hc) out1[(size_t)n*24 + hc] = 0.f;
}

__global__ void k_att_logits(const int* srcAll, const int* dstAll, const unsigned char* status,
                             const float* asrc, const float* adst,
                             float* logits, unsigned* m){
  int e = blockIdx.x*blockDim.x + threadIdx.x;
  if (e >= ETOT) return;
  int s = srcAll[e], d = dstAll[e];
  bool keep = (e >= BB*EE) || (!status[d]);
#pragma unroll
  for (int h = 0; h < 8; ++h){
    float l;
    if (keep){ float z = asrc[(size_t)s*8+h] + adst[(size_t)d*8+h]; l = lrelu(z, 0.2f); }
    else l = -1e9f;
    logits[(size_t)e*8+h] = l;
    atomicMax(&m[(size_t)d*8+h], fflip(l));
  }
}

__global__ void k_att_exp(const int* dstAll, const unsigned char* status,
                          const float* logits, const unsigned* m,
                          float* ebuf, float* ssum){
  int e = blockIdx.x*blockDim.x + threadIdx.x;
  if (e >= ETOT) return;
  int d = dstAll[e];
  bool keep = (e >= BB*EE) || (!status[d]);
#pragma unroll
  for (int h = 0; h < 8; ++h){
    float ev = 0.f;
    if (keep) ev = expf(logits[(size_t)e*8+h] - funflip(m[(size_t)d*8+h]));
    ebuf[(size_t)e*8+h] = ev;
    if (ev != 0.f) atomicAdd(&ssum[(size_t)d*8+h], ev);
  }
}

__global__ void k_att_agg(const int* srcAll, const int* dstAll,
                          const float* ebuf, const float* ssum, const float* hsrc,
                          int C, float* outAcc, float* alphaOut){
  int e = blockIdx.x*blockDim.x + threadIdx.x;
  if (e >= ETOT) return;
  int s = srcAll[e], d = dstAll[e];
  for (int h = 0; h < 8; ++h){
    float ev = ebuf[(size_t)e*8+h];
    float alpha = ev / (ssum[(size_t)d*8+h] + 1e-16f);
    if (alphaOut) alphaOut[(size_t)e*8+h] = alpha;
    if (alpha != 0.f){
      for (int c = 0; c < C; ++c)
        atomicAdd(&outAcc[(size_t)d*(8*C) + h*C + c],
                  hsrc[(size_t)s*(8*C) + h*C + c] * alpha);
    }
  }
}

__global__ void k_finish1(const float* out1, const float* g1b, h16* h1_16,
                          unsigned* m2, float* s2, float* out2){
  int n = blockIdx.x*blockDim.x + threadIdx.x;
  if (n >= BNN) return;
#pragma unroll
  for (int hc = 0; hc < 24; ++hc){
    float v = lrelu(out1[(size_t)n*24+hc] + g1b[hc], 0.01f);
    h1_16[(size_t)n*32+hc] = (h16)v;
  }
#pragma unroll
  for (int k = 24; k < 32; ++k) h1_16[(size_t)n*32+k] = (h16)0.f;  // K padding
#pragma unroll
  for (int h = 0; h < 8; ++h){ m2[n*8+h] = fflip(-3.0e38f); s2[n*8+h] = 0.f; }
#pragma unroll
  for (int j = 0; j < 16; ++j) out2[(size_t)n*16+j] = 0.f;
}

__global__ void k_prep2(const float* h2h, const float* g2as, const float* g2ad,
                        float* asrc2, float* adst2){
  int n = blockIdx.x*blockDim.x + threadIdx.x;
  if (n >= BNN) return;
#pragma unroll
  for (int h = 0; h < 8; ++h){
    float a = h2h[(size_t)n*16 + h*2 + 0], b = h2h[(size_t)n*16 + h*2 + 1];
    asrc2[n*8+h] = a*g2as[h*2+0] + b*g2as[h*2+1];
    adst2[n*8+h] = a*g2ad[h*2+0] + b*g2ad[h*2+1];
  }
}

__global__ void k_finish2(const float* out2, const float* g2b, h16* hflat16){
  int n = blockIdx.x*blockDim.x + threadIdx.x;
  if (n >= BNN) return;
  int b = n / NN, i = n % NN;
#pragma unroll
  for (int c = 0; c < 2; ++c){
    float acc = 0.f;
#pragma unroll
    for (int h = 0; h < 8; ++h) acc += out2[(size_t)n*16 + h*2 + c];
    float v = lrelu(acc*(1.f/8.f) + g2b[c], 0.01f);
    hflat16[(size_t)b*2016 + 2*i + c] = (h16)v;
  }
  if (i < 16) hflat16[(size_t)b*2016 + 2000 + i] = (h16)0.f;       // K padding
}

// ---------------- host launcher ----------------

extern "C" void kernel_launch(void* const* d_in, const int* in_sizes, int n_in,
                              void* d_out, int out_size, void* d_ws, size_t ws_size,
                              hipStream_t stream) {
  (void)in_sizes; (void)n_in; (void)out_size; (void)ws_size;
  const float* rain  = (const float*)d_in[0];
  const float* infl  = (const float*)d_in[1];
  const int*   eidx  = (const int*)  d_in[2];
  const float* Wih   = (const float*)d_in[3];
  const float* Whh   = (const float*)d_in[4];
  const float* bih   = (const float*)d_in[5];
  const float* bhh   = (const float*)d_in[6];
  const float* fcw   = (const float*)d_in[7];
  const float* fcb   = (const float*)d_in[8];
  const float* g1w   = (const float*)d_in[9];
  const float* g1as  = (const float*)d_in[10];
  const float* g1ad  = (const float*)d_in[11];
  const float* g1b   = (const float*)d_in[12];
  const float* g2w   = (const float*)d_in[13];
  const float* g2as  = (const float*)d_in[14];
  const float* g2ad  = (const float*)d_in[15];
  const float* g2b   = (const float*)d_in[16];
  const float* l1w   = (const float*)d_in[17];
  const float* l1b   = (const float*)d_in[18];
  const float* l2w   = (const float*)d_in[19];
  const float* l2b   = (const float*)d_in[20];

  float* outF    = (float*)d_out;
  float* pred    = outF;                               // [B,T,2N]
  float* lateral = outF + (size_t)BB*TT*2*NN;          // [B,T,N]
  float* alphas  = lateral + (size_t)BB*TT*NN;         // [T,Etot,8]

  char* W = (char*)d_ws; size_t off = 0;
  auto take = [&](size_t bytes)->char*{
    char* p = W + off; off = (off + bytes + 255) & ~(size_t)255; return p;
  };
  int*      srcAll  = (int*)     take((size_t)ETOT*4);
  int*      dstAll  = (int*)     take((size_t)ETOT*4);
  h16*      hs16    = (h16*)     take((size_t)2048*64*2);
  h16*      fc16    = (h16*)     take((size_t)1008*64*2);
  h16*      l1w16   = (h16*)     take((size_t)256*2016*2);
  h16*      l2w16   = (h16*)     take((size_t)2000*256*2);
  h16*      g2w16   = (h16*)     take((size_t)16*32*2);
  float*    runoff  = (float*)   take((size_t)2048*1000*4);
  float*    xn      = (float*)   take((size_t)BNN*2*4);
  float*    xbuf    = (float*)   take((size_t)BNN*3*4);
  unsigned char* status = (unsigned char*)take((size_t)BNN);
  float*    hg1     = (float*)   take((size_t)BNN*24*4);
  float*    asrc1   = (float*)   take((size_t)BNN*8*4);
  float*    adst1   = (float*)   take((size_t)BNN*8*4);
  unsigned* m1      = (unsigned*)take((size_t)BNN*8*4);
  float*    s1      = (float*)   take((size_t)BNN*8*4);
  float*    out1    = (float*)   take((size_t)BNN*24*4);
  h16*      h1_16   = (h16*)     take((size_t)BNN*32*2);
  float*    h2h     = (float*)   take((size_t)BNN*16*4);
  float*    asrc2   = (float*)   take((size_t)BNN*8*4);
  float*    adst2   = (float*)   take((size_t)BNN*8*4);
  unsigned* m2      = (unsigned*)take((size_t)BNN*8*4);
  float*    s2      = (float*)   take((size_t)BNN*8*4);
  float*    logitsB = (float*)   take((size_t)ETOT*8*4);
  float*    e2      = (float*)   take((size_t)ETOT*8*4);
  float*    out2    = (float*)   take((size_t)BNN*16*4);
  h16*      hflat16 = (h16*)     take((size_t)32*2016*2);
  h16*      hn16    = (h16*)     take((size_t)32*256*2);

  // ---- one-time setup ----
  k_build_edges<<<(ETOT+255)/256, 256, 0, stream>>>(eidx, srcAll, dstAll);
  k_f32_to_f16_pad<<<(1008*64+255)/256, 256, 0, stream>>>(fcw, fc16, 1000, 64, 64, 1008);
  k_f32_to_f16_pad<<<(256*2016+255)/256, 256, 0, stream>>>(l1w, l1w16, 256, 2000, 2016, 256);
  k_f32_to_f16_pad<<<(2000*256+255)/256, 256, 0, stream>>>(l2w, l2w16, 2000, 256, 256, 2000);
  k_f32_to_f16_pad<<<(16*32+255)/256, 256, 0, stream>>>(g2w, g2w16, 16, 24, 32, 16);
  k_zero_f32<<<(BNN*2+255)/256, 256, 0, stream>>>(xn, BNN*2);
  k_lstm<<<1, 256, 0, stream>>>(rain, Wih, Whh, bih, bhh, hs16);
  // runoff = leaky_relu(hs @ fc_wT + fc_b): M=2048(128 tiles), N=1008(63 tiles), K=64
  k_wmma_gemm<<<(128*63+7)/8, 256, 0, stream>>>(hs16, 64, fc16, 64, fcb,
      128, 63, 64, 1000, 1, runoff, 1000, nullptr, 0, nullptr, 0);

  // ---- sequential scan over T ----
  for (int t = 0; t < TT; ++t){
    float* alphaT = alphas + (size_t)t*ETOT*8;
    k_prep1<<<125, 256, 0, stream>>>(runoff, infl, xn, g1w, g1as, g1ad, t,
        lateral, xbuf, status, hg1, asrc1, adst1, m1, s1, out1);
    // GAT layer 1 attention (alpha written to output slab)
    k_att_logits<<<375, 256, 0, stream>>>(srcAll, dstAll, status, asrc1, adst1, logitsB, m1);
    k_att_exp   <<<375, 256, 0, stream>>>(dstAll, status, logitsB, m1, alphaT, s1);
    k_att_agg   <<<375, 256, 0, stream>>>(srcAll, dstAll, alphaT, s1, hg1, 3, out1, alphaT);
    k_finish1   <<<125, 256, 0, stream>>>(out1, g1b, h1_16, m2, s2, out2);
    // h2h = h1 @ g2_wT : M=32000(2000 tiles), N=16(1 tile), K=32
    k_wmma_gemm<<<(2000+7)/8, 256, 0, stream>>>(h1_16, 32, g2w16, 32, nullptr,
        2000, 1, 32, 16, 0, h2h, 16, nullptr, 0, nullptr, 0);
    // GAT layer 2 attention
    k_prep2     <<<125, 256, 0, stream>>>(h2h, g2as, g2ad, asrc2, adst2);
    k_att_logits<<<375, 256, 0, stream>>>(srcAll, dstAll, status, asrc2, adst2, logitsB, m2);
    k_att_exp   <<<375, 256, 0, stream>>>(dstAll, status, logitsB, m2, e2, s2);
    k_att_agg   <<<375, 256, 0, stream>>>(srcAll, dstAll, e2, s2, h2h, 2, out2, nullptr);
    k_finish2   <<<125, 256, 0, stream>>>(out2, g2b, hflat16);
    // hn = leaky_relu(hflat @ l1_wT + l1_b): M=32(2), N=256(16), K=2016 -> f16 mirror only
    k_wmma_gemm<<<(2*16+7)/8, 256, 0, stream>>>(hflat16, 2016, l1w16, 2016, l1b,
        2, 16, 2016, 256, 1, nullptr, 0, nullptr, 0, hn16, 256);
    // xn' = softplus(hn @ l2_wT + l2_b): M=32(2), N=2000(125), K=256; dual store xn + prediction
    k_wmma_gemm<<<(2*125+7)/8, 256, 0, stream>>>(hn16, 256, l2w16, 256, l2b,
        2, 125, 256, 2000, 2, xn, 2000, pred + (size_t)t*2*NN, TT*2*NN, nullptr, 0);
  }
}
